// CrossAttention_6837587935843
// MI455X (gfx1250) — matrix operand order repaired and checked by reading, not compiled, not run on hardware
//
#include <hip/hip_runtime.h>
#include <cstdint>
#include <cstddef>

// ---------------------------------------------------------------------------
// CDNA5 (gfx1250) cross-attention block.  All matmuls via
// v_wmma_f32_16x16x32_bf16 (wave32 fragments per CDNA5 ISA).
// ---------------------------------------------------------------------------

typedef __attribute__((ext_vector_type(16))) __bf16 v16bf;
typedef __attribute__((ext_vector_type(8)))  float  v8f;

struct __align__(16) U4 { unsigned int x, y, z, w; };
struct __align__(8)  U2 { unsigned int x, y; };
struct __align__(16) F4 { float x, y, z, w; };

union FragAB { v16bf v; U4 q[2]; };

constexpr int BN   = 2;
constexpr int SEQ  = 2048;   // N == M
constexpr int DIM  = 1024;
constexpr int NH   = 16;
constexpr int HDIM = 64;
constexpr int FF   = 4096;

__device__ inline unsigned short f2bf(float f) {
  unsigned int u = __float_as_uint(f);
  unsigned int r = u + 0x7FFFu + ((u >> 16) & 1u);   // round-to-nearest-even
  return (unsigned short)(r >> 16);
}

__device__ inline v8f wmma_bf16(const FragAB& a, const FragAB& b, v8f c) {
  return __builtin_amdgcn_wmma_f32_16x16x32_bf16(
      /*neg_a=*/false, a.v, /*neg_b=*/false, b.v,
      /*c_mod=*/(short)0, c, /*reuse_a=*/false, /*reuse_b=*/false);
}

// ---------------------------------------------------------------------------
// f32 -> bf16 conversion (4 elems / thread)
// ---------------------------------------------------------------------------
__global__ void cvt_bf16_kernel(const float* __restrict__ in,
                                unsigned short* __restrict__ out, int n4) {
  int i = blockIdx.x * 256 + threadIdx.x;
  if (i < n4) {
    F4 f = ((const F4*)in)[i];
    union { unsigned short s[4]; U2 u; } o;
    o.s[0] = f2bf(f.x); o.s[1] = f2bf(f.y);
    o.s[2] = f2bf(f.z); o.s[3] = f2bf(f.w);
    ((U2*)out)[i] = o.u;
  }
}

// ---------------------------------------------------------------------------
// Tiled bf16 GEMM with WMMA.  C[rows x cols] = A[rows x K] * B[K x cols] + bias
// 128x128 tile per 256-thread block (8 waves, 2x4), K-step 32.
// EPI 0: f32 out, head-major [B,H,T,64] scatter (QKV projections)
// EPI 1: gelu -> bf16 row-major (fc1)
// EPI 2: f32 out = acc + bias + residual (fc2, writes d_out)
// ---------------------------------------------------------------------------
template<int EPI>
__global__ __launch_bounds__(256) void gemm_bf16_kernel(
    const unsigned short* __restrict__ A,
    const unsigned short* __restrict__ Bw,
    const float* __restrict__ bias,
    const float* __restrict__ resid,
    float* __restrict__ outF,
    unsigned short* __restrict__ outB,
    int K, int cols)
{
  constexpr int TM = 128, TN = 128, TK = 32, LD = TK + 8;  // LD*2B = 80 (16B aligned)
  __shared__ __align__(16) unsigned short As[TM * LD];
  __shared__ __align__(16) unsigned short Bs[TN * LD];     // transposed: [col][k]

  const int tid  = threadIdx.x;
  const int lane = tid & 31;
  const int wave = tid >> 5;
  const int wm   = wave & 1;    // 2 waves along M
  const int wn   = wave >> 1;   // 4 waves along N
  const int al   = lane & 15;
  const int kh   = lane >> 4;
  const int row0 = blockIdx.y * TM;
  const int col0 = blockIdx.x * TN;

  v8f acc[4][2];
  #pragma unroll
  for (int i = 0; i < 4; i++)
    #pragma unroll
    for (int j = 0; j < 2; j++)
      #pragma unroll
      for (int e = 0; e < 8; e++) acc[i][j][e] = 0.f;

  const int arow = tid >> 1, aseg = tid & 1;   // A: 128 rows x 32 k
  const int bk   = tid & 31, bseg = tid >> 5;  // B: 32 k x 128 n

  for (int kk = 0; kk < K; kk += TK) {
    // stage A tile (row-major)
    const U4* ag = (const U4*)(A + (size_t)(row0 + arow) * K + kk + aseg * 16);
    U4 a0 = ag[0], a1 = ag[1];
    *(U4*)(As + arow * LD + aseg * 16)     = a0;
    *(U4*)(As + arow * LD + aseg * 16 + 8) = a1;
    // stage B tile transposed -> Bs[col][k]
    const U4* bg = (const U4*)(Bw + (size_t)(kk + bk) * cols + col0 + bseg * 16);
    union { U4 q[2]; unsigned short s[16]; } bt;
    bt.q[0] = bg[0]; bt.q[1] = bg[1];
    #pragma unroll
    for (int jj = 0; jj < 16; jj++)
      Bs[(bseg * 16 + jj) * LD + bk] = bt.s[jj];
    __syncthreads();

    FragAB af[4], bf[2];
    #pragma unroll
    for (int mi = 0; mi < 4; mi++) {
      const unsigned short* p = As + (wm * 64 + mi * 16 + al) * LD;
      af[mi].q[0] = *(const U4*)(p + 8 * kh);        // k = 8kh..8kh+7
      af[mi].q[1] = *(const U4*)(p + 16 + 8 * kh);   // k = 16+8kh..
    }
    #pragma unroll
    for (int ni = 0; ni < 2; ni++) {
      const unsigned short* p = Bs + (wn * 32 + ni * 16 + al) * LD;
      bf[ni].q[0] = *(const U4*)(p + 8 * kh);        // k = 8kh..8kh+7
      bf[ni].q[1] = *(const U4*)(p + 16 + 8 * kh);   // k = 16+8kh..
    }
    #pragma unroll
    for (int mi = 0; mi < 4; mi++)
      #pragma unroll
      for (int ni = 0; ni < 2; ni++)
        acc[mi][ni] = wmma_bf16(af[mi], bf[ni], acc[mi][ni]);
    __syncthreads();
  }

  // epilogue (C-frag: row = r + 8*kh, col = lane&15)
  #pragma unroll
  for (int mi = 0; mi < 4; mi++) {
    #pragma unroll
    for (int ni = 0; ni < 2; ni++) {
      const int gc = col0 + wn * 32 + ni * 16 + al;
      const float bv = bias[gc];
      #pragma unroll
      for (int r = 0; r < 8; r++) {
        const int gr = row0 + wm * 64 + mi * 16 + 8 * kh + r;
        float v = acc[mi][ni][r] + bv;
        if (EPI == 0) {
          const int bb = gr >> 11, t = gr & (SEQ - 1);
          const int hh = gc >> 6,  hd = gc & (HDIM - 1);
          outF[((size_t)(bb * NH + hh) * SEQ + t) * HDIM + hd] = v;
        } else if (EPI == 1) {
          float g = 0.5f * v * (1.0f + erff(v * 0.70710678118f));
          outB[(size_t)gr * cols + gc] = f2bf(g);
        } else {
          outF[(size_t)gr * cols + gc] = v + resid[(size_t)gr * cols + gc];
        }
      }
    }
  }
}

// ---------------------------------------------------------------------------
// Per-head LayerNorm over 64 elems (one wave per row) -> bf16
// ---------------------------------------------------------------------------
__global__ __launch_bounds__(256) void ln_head_kernel(
    const float* __restrict__ in, const float* __restrict__ w,
    const float* __restrict__ bb, unsigned short* __restrict__ out)
{
  const int row  = blockIdx.x * 8 + (threadIdx.x >> 5);
  const int lane = threadIdx.x & 31;
  const float* p = in + (size_t)row * HDIM;
  float x0 = p[lane], x1 = p[lane + 32];
  float s = x0 + x1;
  #pragma unroll
  for (int m = 16; m >= 1; m >>= 1) s += __shfl_xor(s, m, 32);
  const float mu = s * (1.0f / HDIM);
  const float d0 = x0 - mu, d1 = x1 - mu;
  float vv = d0 * d0 + d1 * d1;
  #pragma unroll
  for (int m = 16; m >= 1; m >>= 1) vv += __shfl_xor(vv, m, 32);
  const float rstd = rsqrtf(vv * (1.0f / HDIM) + 1e-5f);
  out[(size_t)row * HDIM + lane]      = f2bf(d0 * rstd * w[lane] + bb[lane]);
  out[(size_t)row * HDIM + lane + 32] = f2bf(d1 * rstd * w[lane + 32] + bb[lane + 32]);
}

// ---------------------------------------------------------------------------
// Row LayerNorm over DIM=1024 (one block per row) -> bf16
// ---------------------------------------------------------------------------
__global__ __launch_bounds__(256) void ln_row_kernel(
    const float* __restrict__ in, const float* __restrict__ w,
    const float* __restrict__ bb, unsigned short* __restrict__ out)
{
  __shared__ float red[8];
  const int row = blockIdx.x, tid = threadIdx.x;
  const float* p = in + (size_t)row * DIM;
  float x[4];
  #pragma unroll
  for (int i = 0; i < 4; i++) x[i] = p[tid + 256 * i];
  float s = x[0] + x[1] + x[2] + x[3];
  #pragma unroll
  for (int m = 16; m >= 1; m >>= 1) s += __shfl_xor(s, m, 32);
  if ((tid & 31) == 0) red[tid >> 5] = s;
  __syncthreads();
  float tot = 0.f;
  #pragma unroll
  for (int i = 0; i < 8; i++) tot += red[i];
  const float mu = tot * (1.0f / DIM);
  __syncthreads();
  float vv = 0.f;
  #pragma unroll
  for (int i = 0; i < 4; i++) { float d = x[i] - mu; vv += d * d; }
  #pragma unroll
  for (int m = 16; m >= 1; m >>= 1) vv += __shfl_xor(vv, m, 32);
  if ((tid & 31) == 0) red[tid >> 5] = vv;
  __syncthreads();
  float vt = 0.f;
  #pragma unroll
  for (int i = 0; i < 8; i++) vt += red[i];
  const float rstd = rsqrtf(vt * (1.0f / DIM) + 1e-5f);
  #pragma unroll
  for (int i = 0; i < 4; i++) {
    int c = tid + 256 * i;
    out[(size_t)row * DIM + c] = f2bf((x[i] - mu) * rstd * w[c] + bb[c]);
  }
}

// ---------------------------------------------------------------------------
// V f32 [BH][T][64] -> bf16 transposed [BH][64][T]
// ---------------------------------------------------------------------------
__global__ void v_trans_kernel(const float* __restrict__ vr,
                               unsigned short* __restrict__ vt) {
  const size_t i = (size_t)blockIdx.x * 256 + threadIdx.x;  // over BH*64*SEQ
  const int t    = (int)(i & (SEQ - 1));
  const int rest = (int)(i >> 11);
  const int hd   = rest & (HDIM - 1);
  const int bh   = rest >> 6;
  vt[i] = f2bf(vr[((size_t)bh * SEQ + t) * HDIM + hd]);
}

// ---------------------------------------------------------------------------
// Flash attention: grid (B*H, SEQ/128); 8 waves, 16 query rows each.
// Qb,Kb: bf16 [BH][T][64] (post per-head LN); Vt: bf16 [BH][64][T].
// out = softmax(QK^T/8) V + query residual, written into [B,T,DIM] f32.
// ---------------------------------------------------------------------------
__global__ __launch_bounds__(256) void attn_kernel(
    const unsigned short* __restrict__ Qb,
    const unsigned short* __restrict__ Kb,
    const unsigned short* __restrict__ Vt,
    const float* __restrict__ query,
    float* __restrict__ outres)
{
  constexpr int LK = 72;                        // halfword stride (144B, 16B-aligned)
  __shared__ __align__(16) unsigned short Ks[64 * LK];      // [key][hd]
  __shared__ __align__(16) unsigned short Vs[64 * LK];      // [hd][key]
  __shared__ __align__(16) unsigned short Ps[8 * 16 * LK];  // per-wave P staging
  const int bh = blockIdx.x;
  const int b  = bh >> 4, h = bh & 15;
  const int tid = threadIdx.x, lane = tid & 31, wave = tid >> 5;
  const int al = lane & 15, kh = lane >> 4;
  const int q0 = blockIdx.y * 128 + wave * 16;

  // Q fragments (A-layout): row = lane&15, k chunks {8kh..}, {16+8kh..}
  FragAB qf[2];
  {
    const unsigned short* qp = Qb + ((size_t)bh * SEQ + q0 + al) * HDIM;
    qf[0].q[0] = *(const U4*)(qp + 8 * kh);
    qf[0].q[1] = *(const U4*)(qp + 16 + 8 * kh);
    qf[1].q[0] = *(const U4*)(qp + 32 + 8 * kh);
    qf[1].q[1] = *(const U4*)(qp + 48 + 8 * kh);
  }

  float m_i[8], l_i[8];
  v8f Ofr[4];
  #pragma unroll
  for (int r = 0; r < 8; r++) { m_i[r] = -1e30f; l_i[r] = 0.f; }
  #pragma unroll
  for (int n = 0; n < 4; n++)
    #pragma unroll
    for (int e = 0; e < 8; e++) Ofr[n][e] = 0.f;

  const int ldk = tid >> 2, lseg = tid & 3;     // cooperative 64x64 tile loads

  for (int j = 0; j < SEQ; j += 64) {
    {
      const U4* kg = (const U4*)(Kb + ((size_t)bh * SEQ + j + ldk) * HDIM + lseg * 16);
      U4 k0 = kg[0], k1 = kg[1];
      *(U4*)(Ks + ldk * LK + lseg * 16)     = k0;
      *(U4*)(Ks + ldk * LK + lseg * 16 + 8) = k1;
      const U4* vg = (const U4*)(Vt + ((size_t)bh * HDIM + ldk) * SEQ + j + lseg * 16);
      U4 v0 = vg[0], v1 = vg[1];
      *(U4*)(Vs + ldk * LK + lseg * 16)     = v0;
      *(U4*)(Vs + ldk * LK + lseg * 16 + 8) = v1;
    }
    __syncthreads();

    // S = Q K^T (B-frag: col = key = ni*16+al; k = hd, contiguous in Ks row)
    v8f sf[4];
    #pragma unroll
    for (int ni = 0; ni < 4; ni++) {
      FragAB b0, b1;
      const unsigned short* kp = Ks + (ni * 16 + al) * LK;
      b0.q[0] = *(const U4*)(kp + 8 * kh);
      b0.q[1] = *(const U4*)(kp + 16 + 8 * kh);
      b1.q[0] = *(const U4*)(kp + 32 + 8 * kh);
      b1.q[1] = *(const U4*)(kp + 48 + 8 * kh);
      v8f z;
      #pragma unroll
      for (int e = 0; e < 8; e++) z[e] = 0.f;
      z = wmma_bf16(qf[0], b0, z);
      z = wmma_bf16(qf[1], b1, z);
      sf[ni] = z;
    }

    // online softmax; rows live in 16-lane halves -> shfl_xor 1/2/4/8
    #pragma unroll
    for (int r = 0; r < 8; r++) {
      #pragma unroll
      for (int ni = 0; ni < 4; ni++) sf[ni][r] *= 0.125f;   // 1/sqrt(64)
      float mt = fmaxf(fmaxf(sf[0][r], sf[1][r]), fmaxf(sf[2][r], sf[3][r]));
      #pragma unroll
      for (int msk = 1; msk < 16; msk <<= 1) mt = fmaxf(mt, __shfl_xor(mt, msk, 32));
      const float mnew  = fmaxf(m_i[r], mt);
      const float alpha = __expf(m_i[r] - mnew);
      float rs = 0.f;
      #pragma unroll
      for (int ni = 0; ni < 4; ni++) {
        float pv = __expf(sf[ni][r] - mnew);
        sf[ni][r] = pv;
        rs += pv;
      }
      #pragma unroll
      for (int msk = 1; msk < 16; msk <<= 1) rs += __shfl_xor(rs, msk, 32);
      l_i[r] = l_i[r] * alpha + rs;
      m_i[r] = mnew;
      #pragma unroll
      for (int n2 = 0; n2 < 4; n2++) Ofr[n2][r] *= alpha;
    }

    // restage P: C-layout -> LDS -> A-layout (per-wave region; DS in-order)
    unsigned short* pp = Ps + wave * 16 * LK;
    #pragma unroll
    for (int r = 0; r < 8; r++)
      #pragma unroll
      for (int ni = 0; ni < 4; ni++)
        pp[(r + 8 * kh) * LK + ni * 16 + al] = f2bf(sf[ni][r]);

    FragAB pf[2];
    {
      const unsigned short* pl = pp + al * LK;
      pf[0].q[0] = *(const U4*)(pl + 8 * kh);
      pf[0].q[1] = *(const U4*)(pl + 16 + 8 * kh);
      pf[1].q[0] = *(const U4*)(pl + 32 + 8 * kh);
      pf[1].q[1] = *(const U4*)(pl + 48 + 8 * kh);
    }

    // O += P V  (B-frag: col = hd = n2*16+al; k = key, contiguous in Vs row)
    #pragma unroll
    for (int n2 = 0; n2 < 4; n2++) {
      FragAB b0, b1;
      const unsigned short* vp = Vs + (n2 * 16 + al) * LK;
      b0.q[0] = *(const U4*)(vp + 8 * kh);
      b0.q[1] = *(const U4*)(vp + 16 + 8 * kh);
      b1.q[0] = *(const U4*)(vp + 32 + 8 * kh);
      b1.q[1] = *(const U4*)(vp + 48 + 8 * kh);
      Ofr[n2] = wmma_bf16(pf[0], b0, Ofr[n2]);
      Ofr[n2] = wmma_bf16(pf[1], b1, Ofr[n2]);
    }
    __syncthreads();
  }

  // normalize, add residual, scatter heads back into [B,T,DIM]
  #pragma unroll
  for (int n2 = 0; n2 < 4; n2++) {
    const int d = h * HDIM + n2 * 16 + al;
    #pragma unroll
    for (int r = 0; r < 8; r++) {
      const int t = q0 + r + 8 * kh;
      const size_t o = ((size_t)b * SEQ + t) * DIM + d;
      outres[o] = Ofr[n2][r] / l_i[r] + query[o];
    }
  }
}

// ---------------------------------------------------------------------------
// Host launcher
// ---------------------------------------------------------------------------
extern "C" void kernel_launch(void* const* d_in, const int* in_sizes, int n_in,
                              void* d_out, int out_size, void* d_ws, size_t ws_size,
                              hipStream_t stream)
{
  (void)in_sizes; (void)n_in; (void)out_size; (void)ws_size;
  const float* query   = (const float*)d_in[0];
  const float* context = (const float*)d_in[1];
  const float* Wq_w = (const float*)d_in[4];
  const float* Wq_b = (const float*)d_in[5];
  const float* Wk_w = (const float*)d_in[6];
  const float* Wk_b = (const float*)d_in[7];
  const float* Wv_w = (const float*)d_in[8];
  const float* Wv_b = (const float*)d_in[9];
  const float* qn_w = (const float*)d_in[10];
  const float* qn_b = (const float*)d_in[11];
  const float* kn_w = (const float*)d_in[12];
  const float* kn_b = (const float*)d_in[13];
  const float* ffn_w = (const float*)d_in[14];
  const float* ffn_b = (const float*)d_in[15];
  const float* fc1_b = (const float*)d_in[17];
  const float* fc1_w = (const float*)d_in[16];
  const float* fc2_w = (const float*)d_in[18];
  const float* fc2_b = (const float*)d_in[19];

  char* ws = (char*)d_ws;
  const size_t MB = 1024ull * 1024ull;
  unsigned short* qb   = (unsigned short*)(ws + 0 * MB);    // 8MB query bf16
  unsigned short* cb   = (unsigned short*)(ws + 8 * MB);    // 8MB context bf16
  unsigned short* wqb  = (unsigned short*)(ws + 16 * MB);   // 2MB
  unsigned short* wkb  = (unsigned short*)(ws + 18 * MB);   // 2MB
  unsigned short* wvb  = (unsigned short*)(ws + 20 * MB);   // 2MB
  unsigned short* f1b  = (unsigned short*)(ws + 22 * MB);   // 8MB
  unsigned short* f2b  = (unsigned short*)(ws + 30 * MB);   // 8MB
  float* Qraw = (float*)(ws + 38 * MB);                     // 16MB [BH][T][64]
  float* Kraw = (float*)(ws + 54 * MB);                     // 16MB
  float* Vraw = (float*)(ws + 70 * MB);                     // 16MB
  unsigned short* Qn  = (unsigned short*)(ws + 86 * MB);    // 8MB
  unsigned short* Kn  = (unsigned short*)(ws + 94 * MB);    // 8MB
  unsigned short* Vtb = (unsigned short*)(ws + 102 * MB);   // 8MB [BH][64][T]
  float* attnres = (float*)(ws + 110 * MB);                 // 16MB (..126MB peak)
  unsigned short* xmlp = (unsigned short*)(ws + 38 * MB);   // alias dead Qraw
  unsigned short* h1   = (unsigned short*)(ws + 54 * MB);   // alias dead Kraw+Vraw

  auto cvt = [&](const float* src, unsigned short* dst, int n) {
    int n4 = n / 4;
    cvt_bf16_kernel<<<dim3((n4 + 255) / 256), dim3(256), 0, stream>>>(src, dst, n4);
  };
  cvt(query,   qb,  BN * SEQ * DIM);
  cvt(context, cb,  BN * SEQ * DIM);
  cvt(Wq_w,    wqb, DIM * DIM);
  cvt(Wk_w,    wkb, DIM * DIM);
  cvt(Wv_w,    wvb, DIM * DIM);
  cvt(fc1_w,   f1b, DIM * FF);
  cvt(fc2_w,   f2b, FF * DIM);

  dim3 blk(256);
  const int ROWS = BN * SEQ;  // 4096 tokens

  // QKV projections -> head-major f32
  gemm_bf16_kernel<0><<<dim3(DIM / 128, ROWS / 128), blk, 0, stream>>>(
      qb, wqb, Wq_b, nullptr, Qraw, nullptr, DIM, DIM);
  gemm_bf16_kernel<0><<<dim3(DIM / 128, ROWS / 128), blk, 0, stream>>>(
      cb, wkb, Wk_b, nullptr, Kraw, nullptr, DIM, DIM);
  gemm_bf16_kernel<0><<<dim3(DIM / 128, ROWS / 128), blk, 0, stream>>>(
      cb, wvb, Wv_b, nullptr, Vraw, nullptr, DIM, DIM);

  const int HROWS = BN * NH * SEQ;  // 65536 head-rows
  ln_head_kernel<<<dim3(HROWS / 8), blk, 0, stream>>>(Qraw, qn_w, qn_b, Qn);
  ln_head_kernel<<<dim3(HROWS / 8), blk, 0, stream>>>(Kraw, kn_w, kn_b, Kn);
  v_trans_kernel<<<dim3(BN * NH * HDIM * SEQ / 256), blk, 0, stream>>>(Vraw, Vtb);

  attn_kernel<<<dim3(BN * NH, SEQ / 128), blk, 0, stream>>>(Qn, Kn, Vtb, query, attnres);

  // MLP with pre-norm
  ln_row_kernel<<<dim3(ROWS), blk, 0, stream>>>(attnres, ffn_w, ffn_b, xmlp);
  gemm_bf16_kernel<1><<<dim3(FF / 128, ROWS / 128), blk, 0, stream>>>(
      xmlp, f1b, fc1_b, nullptr, nullptr, h1, DIM, FF);
  gemm_bf16_kernel<2><<<dim3(DIM / 128, ROWS / 128), blk, 0, stream>>>(
      h1, f2b, fc2_b, attnres, (float*)d_out, nullptr, FF, DIM);
}
